// KPN_46145128629027
// MI455X (gfx1250) — matrix operand order repaired and checked by reading, not compiled, not run on hardware
//
#include <hip/hip_runtime.h>
#include <hip/hip_bf16.h>

typedef __attribute__((ext_vector_type(16))) _Float16 v16h;
typedef __attribute__((ext_vector_type(8)))  float    v8f;

#define Hh 384
#define Ww 384
#define HW (384*384)
#define PW 386
#define PHW (386*386)

#define WROW 72                      // padded LDS row stride (f16) -> 144B, 16B aligned, bank-conflict-free
#define LDSW_ELEMS (9 * 64 * WROW)   // 41472
#define LDSX_ELEMS (3 * 130 * WROW)  // 28080
#define SHMEM_BYTES ((LDSW_ELEMS + LDSX_ELEMS) * 2)  // 139104

// ---------------- utility kernels ----------------

__global__ void zero_u4(uint4* __restrict__ p, long n) {
    long i = (long)blockIdx.x * blockDim.x + threadIdx.x;
    long stride = (long)gridDim.x * blockDim.x;
    uint4 z; z.x = 0u; z.y = 0u; z.z = 0u; z.w = 0u;
    for (; i < n; i += stride) p[i] = z;
}

__global__ void pack_in6(const float* __restrict__ src, _Float16* __restrict__ dst) {
    int i = blockIdx.x * 256 + threadIdx.x;
    if (i >= 6 * HW) return;
    int c = i / HW; int r = i - c * HW;
    int y = r / Ww; int x = r - y * Ww;
    dst[(c * PW + y + 1) * PW + (x + 1)] = (_Float16)src[i];
}

// ---------------- WMMA implicit-GEMM conv3x3 ----------------
// Block: 256 threads (8 waves). Computes 64 out-channels x 128 pixels of one row.
// LDS: weights f16 [9][64][WROW] (c fastest) + input tile f16 [3][130][WROW] (c fastest).
// Fragments are 2x ds_load_b128 per operand; K-loop register double-buffered.

union FragU { uint4 q[2]; v16h v; };

template <int CBLKS>
__global__ __launch_bounds__(256) void conv3x3_wmma(
    const _Float16* __restrict__ in, int cin, int cin_planes,
    const float* __restrict__ wgt, const float* __restrict__ bias, int oc,
    _Float16* __restrict__ out_h, float* __restrict__ out_f,
    const _Float16* __restrict__ res, int do_relu)
{
    extern __shared__ _Float16 smem[];
    _Float16* lds_w = smem;               // [pos][o][WROW]
    _Float16* lds_x = smem + LDSW_ELEMS;  // [kh][wl][WROW]

    const int h  = blockIdx.y;
    const int w0 = blockIdx.x * 128;

    // stage weights: linear global index (coalesced for cin=64), guarded + zero-filled
    for (int i = threadIdx.x; i < 9 * 64 * 64; i += 256) {
        int pos = i % 9;
        int rem = i / 9;
        int c = rem & 63;
        int o = rem >> 6;                 // i == ((o*64)+c)*9+pos when cin==64
        float v = 0.f;
        if (o < oc && c < cin) v = wgt[(o * cin + c) * 9 + pos];
        lds_w[(pos * 64 + o) * WROW + c] = (_Float16)v;
    }
    // stage input tile: wl fastest -> contiguous global reads
    for (int i = threadIdx.x; i < 64 * 3 * 130; i += 256) {
        int wl = i % 130;
        int t  = i / 130;
        int kh = t % 3;
        int c  = t / 3;
        _Float16 v = (_Float16)0.f;
        if (c < cin_planes) v = in[(c * PW + (h + kh)) * PW + (w0 + wl)];
        lds_x[(kh * 130 + wl) * WROW + c] = v;
    }
    __syncthreads();

    const int lane = threadIdx.x & 31;
    const int wv   = threadIdx.x >> 5;
    const int m    = lane & 15;      // M row (A) / N col (B,D)
    const int half = lane >> 4;
    const int mt   = wv & 3;         // 16-output-channel tile
    const int p0   = (wv >> 2) * 64; // 64-pixel half owned by this wave

    v8f acc[4];
    {
        v8f z = {0.f, 0.f, 0.f, 0.f, 0.f, 0.f, 0.f, 0.f};
        acc[0] = z; acc[1] = z; acc[2] = z; acc[3] = z;
    }

    auto load_frags = [&](int s, FragU& Af, FragU* Bf) {
        const int pos = s / CBLKS;
        const int cb  = s % CBLKS;
        const int kh  = pos / 3;
        const int kw  = pos - kh * 3;
        // A fragment (16x32 f16, contiguous k = 16*half + s, same order as B)
        const _Float16* abase = &lds_w[(pos * 64 + mt * 16 + m) * WROW + cb * 32];
        Af.q[0] = *(const uint4*)(abase + 16 * half);
        Af.q[1] = *(const uint4*)(abase + 16 * half + 8);
        // B fragments (32x16 f16): lanes 0-15 K=0..15, lanes 16-31 K=16..31
#pragma unroll
        for (int nt = 0; nt < 4; ++nt) {
            const _Float16* bbase =
                &lds_x[(kh * 130 + p0 + nt * 16 + m + kw) * WROW + cb * 32];
            Bf[nt].q[0] = *(const uint4*)(bbase + 16 * half);
            Bf[nt].q[1] = *(const uint4*)(bbase + 16 * half + 8);
        }
    };

    const int STEPS = 9 * CBLKS;
    FragU A[2];
    FragU B[2][4];
    load_frags(0, A[0], B[0]);
#pragma unroll
    for (int s = 0; s < STEPS; ++s) {
        const int cur = s & 1;
        const int nxt = cur ^ 1;
        if (s + 1 < STEPS) load_frags(s + 1, A[nxt], B[nxt]);
#pragma unroll
        for (int nt = 0; nt < 4; ++nt) {
            acc[nt] = __builtin_amdgcn_wmma_f32_16x16x32_f16(
                false, A[cur].v, false, B[cur][nt].v, (short)0, acc[nt], false, false);
        }
    }

    // epilogue: bias (+ residual) (+ relu), store f16 padded or f32 dense
#pragma unroll
    for (int nt = 0; nt < 4; ++nt) {
#pragma unroll
        for (int r = 0; r < 8; ++r) {
            int o = mt * 16 + r + 8 * half;      // D layout: lanes16-31 -> M+8
            if (o >= oc) continue;
            int wpix = w0 + p0 + nt * 16 + m;
            float v = acc[nt][r] + bias[o];
            if (res) v += (float)res[(o * PW + h + 1) * PW + (wpix + 1)];
            if (do_relu) v = fmaxf(v, 0.f);
            if (out_f) out_f[(o * Hh + h) * Ww + wpix] = v;
            else       out_h[(o * PW + h + 1) * PW + (wpix + 1)] = (_Float16)v;
        }
    }
}

// ---------------- KPN radial filter bank ----------------

__device__ __forceinline__ float radial_val(const float* ck, int wide, int a, int b) {
    float d = sqrtf((float)(a * a + b * b));
    float mid = (float)(wide - 1);
    if (d > mid) return 0.f;
    float lo = floorf(d), hi = ceilf(d);
    int li = (int)lo; if (li > wide - 1) li = wide - 1;
    int hi_i = (int)hi; if (hi_i > wide - 1) hi_i = wide - 1;
    float w_hi = d - lo;
    float w_lo = (lo == hi) ? 1.0f : (hi - d);
    return w_lo * ck[li] + w_hi * ck[hi_i];
}

__global__ __launch_bounds__(256) void kpn_filter(
    const float* __restrict__ core, const float* __restrict__ data, float* __restrict__ out)
{
    int p = blockIdx.x * 256 + threadIdx.x;
    if (p >= HW) return;
    int y = p / Ww;
    int x = p - y * Ww;

    // output 0: plain copy of data
    out[0 * HW + p] = data[0 * HW + p];
    out[1 * HW + p] = data[1 * HW + p];
    out[2 * HW + p] = data[2 * HW + p];

    int off = 0;
    float ck[8];
    for (int wide = 2; wide <= 8; ++wide) {
        int mid = wide - 1;
        for (int j = 0; j < wide; ++j) ck[j] = fabsf(core[(off + j) * HW + p]);
        off += wide;

        // pass 1: max over K*K (masked entries are exactly 0, included)
        float mx = -3.0e38f;
        for (int a = -mid; a <= mid; ++a)
            for (int b = -mid; b <= mid; ++b)
                mx = fmaxf(mx, radial_val(ck, wide, a, b));
        // pass 2: sum of exp
        float s = 0.f;
        for (int a = -mid; a <= mid; ++a)
            for (int b = -mid; b <= mid; ++b)
                s += __expf(radial_val(ck, wide, a, b) - mx);
        // pass 3: apply to zero-padded data
        float a0 = 0.f, a1 = 0.f, a2 = 0.f;
        for (int a = -mid; a <= mid; ++a) {
            int yy = y + a;
            for (int b = -mid; b <= mid; ++b) {
                int xx = x + b;
                float e = __expf(radial_val(ck, wide, a, b) - mx);
                if (yy >= 0 && yy < Hh && xx >= 0 && xx < Ww) {
                    int base = yy * Ww + xx;
                    a0 += e * data[base];
                    a1 += e * data[HW + base];
                    a2 += e * data[2 * HW + base];
                }
            }
        }
        float inv = 1.f / s;
        int t = wide - 1; // output tensor index 1..7
        out[(t * 3 + 0) * HW + p] = a0 * inv;
        out[(t * 3 + 1) * HW + p] = a1 * inv;
        out[(t * 3 + 2) * HW + p] = a2 * inv;
    }
}

// ---------------- launcher ----------------

extern "C" void kernel_launch(void* const* d_in, const int* in_sizes, int n_in,
                              void* d_out, int out_size, void* d_ws, size_t ws_size,
                              hipStream_t stream) {
    (void)in_sizes; (void)n_in; (void)out_size; (void)ws_size;
    const float* d_est   = (const float*)d_in[0];
    const float* d_data  = (const float*)d_in[1];
    const float* w_first = (const float*)d_in[2];
    const float* b_first = (const float*)d_in[3];
    const float* w_blk1  = (const float*)d_in[4];
    const float* b_blk1  = (const float*)d_in[5];
    const float* w_blk2  = (const float*)d_in[6];
    const float* b_blk2  = (const float*)d_in[7];
    const float* w_out   = (const float*)d_in[8];
    const float* b_out   = (const float*)d_in[9];
    float* out = (float*)d_out;

    char* ws = (char*)d_ws;
    size_t o1 = (size_t)32 * PHW * 2;           // in6pad f16 [32][386][386]
    size_t o2 = o1 + (size_t)64 * PHW * 2;      // xpad   f16 [64][386][386]
    size_t o3 = o2 + (size_t)64 * PHW * 2;      // ypad   f16 [64][386][386]
    _Float16* in6 = (_Float16*)ws;
    _Float16* xp  = (_Float16*)(ws + o1);
    _Float16* yp  = (_Float16*)(ws + o2);
    float*    core = (float*)(ws + o3);         // f32 [35][384][384]

    // zero all padded f16 staging (borders must be 0; ws may be poisoned)
    long nz = (long)(32 + 64 + 64) * PHW * 2 / 16;
    zero_u4<<<4096, 256, 0, stream>>>((uint4*)ws, nz);

    pack_in6<<<(6 * HW + 255) / 256, 256, 0, stream>>>(d_est, in6);

    hipFuncSetAttribute(reinterpret_cast<const void*>(&conv3x3_wmma<1>),
                        hipFuncAttributeMaxDynamicSharedMemorySize, SHMEM_BYTES);
    hipFuncSetAttribute(reinterpret_cast<const void*>(&conv3x3_wmma<2>),
                        hipFuncAttributeMaxDynamicSharedMemorySize, SHMEM_BYTES);

    dim3 cg(3, 384, 1);
    // first conv: 6 -> 64 (one 32-wide K block)
    conv3x3_wmma<1><<<cg, 256, SHMEM_BYTES, stream>>>(in6, 6, 32, w_first, b_first, 64,
                                                      xp, nullptr, nullptr, 0);
    // three residual blocks
    for (int i = 0; i < 3; ++i) {
        conv3x3_wmma<2><<<cg, 256, SHMEM_BYTES, stream>>>(
            xp, 64, 64, w_blk1 + (size_t)i * 64 * 64 * 9, b_blk1 + i * 64, 64,
            yp, nullptr, nullptr, 1);
        conv3x3_wmma<2><<<cg, 256, SHMEM_BYTES, stream>>>(
            yp, 64, 64, w_blk2 + (size_t)i * 64 * 64 * 9, b_blk2 + i * 64, 64,
            xp, nullptr, xp, 0);
    }
    // output conv: 64 -> 35, f32 dense
    conv3x3_wmma<2><<<cg, 256, SHMEM_BYTES, stream>>>(xp, 64, 64, w_out, b_out, 35,
                                                      nullptr, core, nullptr, 0);

    kpn_filter<<<(HW + 255) / 256, 256, 0, stream>>>(core, d_data, out);
}
